// SkipTransformer_18322330484792
// MI455X (gfx1250) — compile-verified, matrix operand order checked by reading, not run on hardware
//
#include <hip/hip_runtime.h>
#include <cstdint>

typedef __bf16 bf16;
typedef __attribute__((ext_vector_type(16))) __bf16 v16bf;
typedef __attribute__((ext_vector_type(8)))  float  v8f;

#define BB    2
#define NPTS  2048
#define CCH   128
#define DCH   256
#define PHC   64
#define AHC   1024
#define KNN   16
#define NKC   (NPTS*KNN)

// ---------------------------------------------------------------------------
// WMMA GEMM:  Y[b,o,col] = act( sum_c W[o,c] * X[b,c,col] + bias )
// W fp32 row-major; X bf16 (B, x_stride, Ncols) with channel offset.
// SPLIT fuses two K-ranges: c < cinA -> (W1,X1), c >= cinA -> (W2,X2);
// cinA is a multiple of 64, so the per-K-step source select is uniform.
//
// Block: 128 threads = 4 waves; block tile = 16 (Cout) x 128 (cols).
// K-step = 64 (two 32-K sub-tiles per barrier -> 4 WMMAs per sync).
// LDS tiles are stored in *fragment-permuted* order: the 16 bf16 values
// lane L needs for v_wmma_f32_16x16x32_bf16 are contiguous (32B), so each
// fragment is one 32-byte LDS read (2x ds_load_b128).
//   element e of lane L (hs = L>>4):  K = e<8 ? hs*8+e : 16+hs*8+(e-8)
// All per-lane addressing is 32-bit offsets off uniform 64-bit bases
// (avoids per-element v_mul_u64 chains).
// ---------------------------------------------------------------------------
template<bool SPLIT, bool BIAS2, bool RES, bool RELU, bool OUTF, bool OUTH>
__global__ __launch_bounds__(128)
void wmma_gemm_kernel(const float* __restrict__ W1, const float* __restrict__ W2,
                      int cinA, int cinTotal,
                      const float* __restrict__ bias1, const float* __restrict__ bias2,
                      const bf16* __restrict__ X1, int x1s, int x1o,
                      const bf16* __restrict__ X2, int x2s, int x2o,
                      int Cout, int Ncols,
                      float* __restrict__ Yf, bf16* __restrict__ Yh,
                      const float* __restrict__ res)
{
    __shared__ __align__(32) bf16 Alds[1024];   // [u*512  + L*16 + e]
    __shared__ __align__(32) bf16 Blds[8192];   // [u*4096 + (st*32+L)*16 + e]

    const int tid     = threadIdx.x;
    const int lane    = tid & 31;
    const int w       = tid >> 5;          // wave id 0..3
    const int b       = blockIdx.z;
    const int oBase   = blockIdx.y * 16;
    const int colBase = blockIdx.x * 128;
    const int cinB    = cinTotal - cinA;

    // uniform 64-bit base pointers (per-lane offsets stay 32-bit)
    const bf16* X1base = X1 + (size_t)(b * x1s + x1o) * (size_t)Ncols + colBase;
    const bf16* X2base = SPLIT
        ? (X2 + (size_t)(b * x2s + x2o) * (size_t)Ncols + colBase) : nullptr;

    const v8f vzero = {0.f, 0.f, 0.f, 0.f, 0.f, 0.f, 0.f, 0.f};
    v8f acc[2] = {vzero, vzero};

    for (int kk = 0; kk < cinTotal; kk += 64) {
        const bool  second = SPLIT && (kk >= cinA);
        const float* Wb    = second ? W2 : W1;
        const int    wcols = second ? cinB : cinA;
        const bf16*  Xb    = second ? X2base : X1base;
        const int    cb    = second ? (kk - cinA) : kk;

        // --- stage A (16 rows x 64 K) directly into permuted layout
        #pragma unroll
        for (int i = 0; i < 8; ++i) {
            int s  = tid * 8 + i;                       // 0..1023
            int u  = s >> 9;
            int sl = s & 511;
            int L  = sl >> 4;
            int e  = sl & 15;
            int hs = L >> 4;
            int K  = (e < 8) ? (hs * 8 + e) : (16 + hs * 8 + (e - 8));
            uint woff = (uint)(oBase + (L & 15)) * (uint)wcols + (uint)(cb + u * 32 + K);
            Alds[s] = (bf16)Wb[woff];
        }
        // --- stage B (64 K x 128 cols): coalesced 16B global loads,
        //     scatter u16 into permuted fragment layout
        #pragma unroll
        for (int i = 0; i < 8; ++i) {
            int chunk = tid * 8 + i;        // 0..1023 = 64 K-rows x 16 chunks
            int K     = chunk >> 4;         // 0..63
            int col0  = (chunk & 15) * 8;   // 0..120
            uint off  = (uint)(cb + K) * (uint)Ncols + (uint)col0;
            uint4 v4 = *reinterpret_cast<const uint4*>(Xb + off);
            bf16 tmp[8];
            *reinterpret_cast<uint4*>(tmp) = v4;
            int u  = K >> 5;
            int Kl = K & 31;
            int hs, e;
            if (Kl < 16) { hs = Kl >> 3;        e = Kl & 7; }
            else         { hs = (Kl - 16) >> 3; e = 8 + ((Kl - 16) & 7); }
            #pragma unroll
            for (int j = 0; j < 8; ++j) {
                int col = col0 + j;
                Blds[u * 4096 + ((col >> 4) * 32 + hs * 16 + (col & 15)) * 16 + e] = tmp[j];
            }
        }
        __syncthreads();

        // --- 4 WMMAs per barrier; each fragment = one 32B LDS read
        #pragma unroll
        for (int u = 0; u < 2; ++u) {
            v16bf af = *reinterpret_cast<const v16bf*>(&Alds[u * 512 + lane * 16]);
            v16bf b0 = *reinterpret_cast<const v16bf*>(
                           &Blds[u * 4096 + ((2 * w)     * 32 + lane) * 16]);
            v16bf b1 = *reinterpret_cast<const v16bf*>(
                           &Blds[u * 4096 + ((2 * w + 1) * 32 + lane) * 16]);
            acc[0] = __builtin_amdgcn_wmma_f32_16x16x32_bf16(
                         false, af, false, b0, (short)0, acc[0], false, false);
            acc[1] = __builtin_amdgcn_wmma_f32_16x16x32_bf16(
                         false, af, false, b1, (short)0, acc[1], false, false);
        }
        __syncthreads();
    }

    // --- branch-free epilogue; uniform 64-bit base + 32-bit lane offsets
    const int hs = lane >> 4;
    const size_t obase = (size_t)(b * Cout + oBase) * (size_t)Ncols;
    const float* resb = RES  ? res + obase : nullptr;
    float*       Yfb  = OUTF ? Yf  + obase : nullptr;
    bf16*        Yhb  = OUTH ? Yh  + obase : nullptr;
    #pragma unroll
    for (int j = 0; j < 2; ++j) {
        uint co = (uint)colBase + (uint)((2 * w + j) * 16) + (uint)(lane & 15);
        #pragma unroll
        for (int r = 0; r < 8; ++r) {
            int m = r + 8 * hs;
            float vv = acc[j][r] + bias1[oBase + m];
            if (BIAS2) vv += bias2[oBase + m];
            uint oo = (uint)m * (uint)Ncols + co;
            if (RES)  vv += resb[oo];
            if (RELU) vv = fmaxf(vv, 0.f);
            if (OUTF) Yfb[oo] = vv;
            if (OUTH) Yhb[oo] = (bf16)vv;
        }
    }
}

// ---------------------------------------------------------------------------
// Brute-force KNN (include_self), LDS-tiled; per-thread sorted top-16.
// ---------------------------------------------------------------------------
__global__ __launch_bounds__(256)
void knn_kernel(const float* __restrict__ pos, int* __restrict__ idx)
{
    __shared__ float sx[256], sy[256], sz[256];
    int b = blockIdx.y;
    int n = blockIdx.x * 256 + threadIdx.x;
    int t = threadIdx.x;
    float qx = pos[(b * 3 + 0) * NPTS + n];
    float qy = pos[(b * 3 + 1) * NPTS + n];
    float qz = pos[(b * 3 + 2) * NPTS + n];
    float bd[KNN]; int bi[KNN];
    #pragma unroll
    for (int i = 0; i < KNN; ++i) { bd[i] = 3.0e38f; bi[i] = 0; }
    for (int j0 = 0; j0 < NPTS; j0 += 256) {
        __syncthreads();
        sx[t] = pos[(b * 3 + 0) * NPTS + j0 + t];
        sy[t] = pos[(b * 3 + 1) * NPTS + j0 + t];
        sz[t] = pos[(b * 3 + 2) * NPTS + j0 + t];
        __syncthreads();
        for (int jj = 0; jj < 256; ++jj) {
            float dx = qx - sx[jj], dy = qy - sy[jj], dz = qz - sz[jj];
            float d = dx * dx + dy * dy + dz * dz;
            if (d < bd[KNN - 1]) {
                int p = KNN - 1;
                while (p > 0 && d < bd[p - 1]) {
                    bd[p] = bd[p - 1]; bi[p] = bi[p - 1]; --p;
                }
                bd[p] = d; bi[p] = j0 + jj;
            }
        }
    }
    for (int i = 0; i < KNN; ++i)
        idx[((size_t)(b * NPTS + n)) * KNN + i] = bi[i];
}

// ---------------------------------------------------------------------------
// concat(key,query) -> bf16 (B, 2C, N)
// ---------------------------------------------------------------------------
__global__ void concat_kernel(const float* __restrict__ key,
                              const float* __restrict__ query,
                              bf16* __restrict__ xcat)
{
    int i = blockIdx.x * 256 + threadIdx.x;   // B*256*N
    int n = i & (NPTS - 1);
    int c = (i >> 11) & 255;
    int b = i >> 19;
    float v = (c < CCH) ? key  [((size_t)(b * CCH + c)) * NPTS + n]
                        : query[((size_t)(b * CCH + (c - CCH))) * NPTS + n];
    xcat[i] = (bf16)v;
}

// ---------------------------------------------------------------------------
// pos-MLP layer 1: h[b,ch,n,k] = pos_w1(64x3) . pos_rel + b1   (pre-BN, bf16)
// ---------------------------------------------------------------------------
__global__ void posmlp1_kernel(const float* __restrict__ pos,
                               const int* __restrict__ idx,
                               const float* __restrict__ w1,
                               const float* __restrict__ b1,
                               bf16* __restrict__ hpos)
{
    int i = blockIdx.x * 256 + threadIdx.x;   // B*N*K = 65536
    int k = i & 15;
    int n = (i >> 4) & (NPTS - 1);
    int b = i >> 15;
    int jn = idx[((size_t)(b * NPTS + n)) * KNN + k];
    float rx = pos[(b * 3 + 0) * NPTS + n] - pos[(b * 3 + 0) * NPTS + jn];
    float ry = pos[(b * 3 + 1) * NPTS + n] - pos[(b * 3 + 1) * NPTS + jn];
    float rz = pos[(b * 3 + 2) * NPTS + n] - pos[(b * 3 + 2) * NPTS + jn];
    int colidx = n * KNN + k;
    for (int ch = 0; ch < PHC; ++ch) {
        float h = w1[ch * 3] * rx + w1[ch * 3 + 1] * ry + w1[ch * 3 + 2] * rz + b1[ch];
        hpos[((size_t)(b * PHC + ch)) * NKC + colidx] = (bf16)h;
    }
}

// ---------------------------------------------------------------------------
// Per-channel BN stats over (B, n*k): one block per channel -> scale/shift.
// ---------------------------------------------------------------------------
__global__ __launch_bounds__(256)
void bnstats_kernel(const bf16* __restrict__ X, const float* __restrict__ gamma,
                    const float* __restrict__ beta, float* __restrict__ ss,
                    int C, int NK)
{
    __shared__ float rs[256], rs2[256];
    int ch = blockIdx.x, t = threadIdx.x;
    float s = 0.f, s2 = 0.f;
    for (int b = 0; b < BB; ++b) {
        const bf16* p = X + ((size_t)(b * C + ch)) * NK;
        for (int j = t; j < NK; j += 256) {
            float v = (float)p[j];
            s += v; s2 += v * v;
        }
    }
    rs[t] = s; rs2[t] = s2;
    __syncthreads();
    for (int o = 128; o > 0; o >>= 1) {
        if (t < o) { rs[t] += rs[t + o]; rs2[t] += rs2[t + o]; }
        __syncthreads();
    }
    if (t == 0) {
        float M   = (float)(BB * NK);
        float mu  = rs[0] / M;
        float var = rs2[0] / M - mu * mu;
        float sc  = gamma[ch] * rsqrtf(var + 1e-5f);
        ss[ch * 2]     = sc;
        ss[ch * 2 + 1] = beta[ch] - mu * sc;
    }
}

// ---------------------------------------------------------------------------
// In-place BN apply + ReLU on bf16 buffer (B, C, NK)
// ---------------------------------------------------------------------------
__global__ void bnrelu_kernel(bf16* __restrict__ X, const float* __restrict__ ss,
                              int C, int NK, int total)
{
    int i = blockIdx.x * 256 + threadIdx.x;
    if (i >= total) return;
    int ch = (i / NK) % C;
    float v = (float)X[i] * ss[ch * 2] + ss[ch * 2 + 1];
    X[i] = (bf16)fmaxf(v, 0.f);
}

// ---------------------------------------------------------------------------
// s[b,d,n,k] = q[b,d,n] - k[b,d,idx] + pos_emb[b,d,n,k]   (bf16, GEMM input)
// ---------------------------------------------------------------------------
__global__ void qkrel_kernel(const float* __restrict__ qf,
                             const float* __restrict__ kf,
                             const int* __restrict__ idx,
                             const float* __restrict__ posemb,
                             bf16* __restrict__ s)
{
    size_t i = (size_t)blockIdx.x * 256 + threadIdx.x;   // B*D*N*K
    int k = (int)(i & 15);
    int n = (int)((i >> 4) & (NPTS - 1));
    int d = (int)((i >> 15) & (DCH - 1));
    int b = (int)(i >> 23);
    int jn = idx[((size_t)(b * NPTS + n)) * KNN + k];
    size_t pe = ((size_t)(b * DCH + d)) * NKC + (size_t)n * KNN + k;
    float v = qf[((size_t)(b * DCH + d)) * NPTS + n]
            - kf[((size_t)(b * DCH + d)) * NPTS + jn]
            + posemb[pe];
    s[pe] = (bf16)v;
}

// ---------------------------------------------------------------------------
// softmax over K + weighted aggregation:
// agg[b,d,n] = sum_k softmax(e)_k * (v[b,d,n] + pos_emb[b,d,n,k])  -> bf16
// ---------------------------------------------------------------------------
__global__ void softagg_kernel(const float* __restrict__ e,
                               const float* __restrict__ vf,
                               const float* __restrict__ posemb,
                               bf16* __restrict__ agg)
{
    int i = blockIdx.x * 256 + threadIdx.x;   // B*D*N
    int n = i & (NPTS - 1);
    int d = (i >> 11) & (DCH - 1);
    int b = i >> 19;
    size_t base = ((size_t)(b * DCH + d)) * NKC + (size_t)n * KNN;
    float ev[KNN];
    float m = -3.0e38f;
    #pragma unroll
    for (int k = 0; k < KNN; ++k) { ev[k] = e[base + k]; m = fmaxf(m, ev[k]); }
    float sum = 0.f;
    #pragma unroll
    for (int k = 0; k < KNN; ++k) { ev[k] = __expf(ev[k] - m); sum += ev[k]; }
    float vv = vf[((size_t)(b * DCH + d)) * NPTS + n];
    float a = 0.f;
    #pragma unroll
    for (int k = 0; k < KNN; ++k) a += ev[k] * (vv + posemb[base + k]);
    agg[((size_t)(b * DCH + d)) * NPTS + n] = (bf16)(a / sum);
}

// ---------------------------------------------------------------------------
extern "C" void kernel_launch(void* const* d_in, const int* in_sizes, int n_in,
                              void* d_out, int out_size, void* d_ws, size_t ws_size,
                              hipStream_t stream)
{
    const float* pos      = (const float*)d_in[0];
    const float* key      = (const float*)d_in[1];
    const float* query    = (const float*)d_in[2];
    const float* mlp_w1   = (const float*)d_in[3];
    const float* mlp_b1   = (const float*)d_in[4];
    const float* mlp_w2   = (const float*)d_in[5];
    const float* mlp_b2   = (const float*)d_in[6];
    const float* mlp_wsc  = (const float*)d_in[7];
    const float* mlp_bsc  = (const float*)d_in[8];
    const float* wk       = (const float*)d_in[9];
    const float* bk       = (const float*)d_in[10];
    const float* wq       = (const float*)d_in[11];
    const float* bq       = (const float*)d_in[12];
    const float* wv       = (const float*)d_in[13];
    const float* bv       = (const float*)d_in[14];
    const float* pos_w1   = (const float*)d_in[15];
    const float* pos_b1   = (const float*)d_in[16];
    const float* pos_g    = (const float*)d_in[17];
    const float* pos_beta = (const float*)d_in[18];
    const float* pos_w2   = (const float*)d_in[19];
    const float* pos_b2   = (const float*)d_in[20];
    const float* attn_w1  = (const float*)d_in[21];
    const float* attn_b1  = (const float*)d_in[22];
    const float* attn_g   = (const float*)d_in[23];
    const float* attn_beta= (const float*)d_in[24];
    const float* attn_w2  = (const float*)d_in[25];
    const float* attn_b2  = (const float*)d_in[26];
    const float* end_w    = (const float*)d_in[27];
    const float* end_b    = (const float*)d_in[28];
    float* out = (float*)d_out;

    // ---- workspace carve-up (256B-aligned regions)
    char* ws = (char*)d_ws;
    size_t off = 0;
    auto alloc = [&](size_t bytes) -> char* {
        char* p = ws + off;
        off = (off + bytes + 255) & ~(size_t)255;
        return p;
    };
    bf16*  xcat     = (bf16*) alloc((size_t)BB * 256 * NPTS * 2);
    bf16*  h1       = (bf16*) alloc((size_t)BB * CCH * NPTS * 2);
    float* value_f  = (float*)alloc((size_t)BB * CCH * NPTS * 4);
    bf16*  value_bf = (bf16*) alloc((size_t)BB * CCH * NPTS * 2);
    float* kf       = (float*)alloc((size_t)BB * DCH * NPTS * 4);
    float* qf       = (float*)alloc((size_t)BB * DCH * NPTS * 4);
    float* vf       = (float*)alloc((size_t)BB * DCH * NPTS * 4);
    int*   idx      = (int*)  alloc((size_t)BB * NPTS * KNN * 4);
    bf16*  hpos     = (bf16*) alloc((size_t)BB * PHC * NKC * 2);
    float* posemb   = (float*)alloc((size_t)BB * DCH * NKC * 4);
    bf16*  s_bf     = (bf16*) alloc((size_t)BB * DCH * NKC * 2);
    bf16*  a_bf     = (bf16*) alloc((size_t)BB * AHC * NKC * 2);
    float* e_f      = (float*)alloc((size_t)BB * DCH * NKC * 4);
    bf16*  agg_bf   = (bf16*) alloc((size_t)BB * DCH * NPTS * 2);
    float* ss_pos   = (float*)alloc((size_t)PHC * 2 * 4);
    float* ss_attn  = (float*)alloc((size_t)AHC * 2 * 4);

    dim3 blk256(256);
    dim3 blkG(128);

    // 1) concat key/query -> bf16
    concat_kernel<<<dim3((BB * 256 * NPTS) / 256), blk256, 0, stream>>>(key, query, xcat);

    // 2) KNN indices
    knn_kernel<<<dim3(NPTS / 256, BB), blk256, 0, stream>>>(pos, idx);

    // 3) h1 = relu(mlp_w1 . xcat + b1)            [OUTH, RELU]
    wmma_gemm_kernel<false,false,false,true,false,true>
        <<<dim3(NPTS / 128, CCH / 16, BB), blkG, 0, stream>>>(
        mlp_w1, nullptr, 2 * CCH, 2 * CCH, mlp_b1, nullptr,
        xcat, 2 * CCH, 0, nullptr, 0, 0,
        CCH, NPTS, nullptr, h1, nullptr);

    // 4) value = mlp_w2.h1 + b2 + mlp_wsc.xcat + bsc   [SPLIT, BIAS2, OUTF+OUTH]
    wmma_gemm_kernel<true,true,false,false,true,true>
        <<<dim3(NPTS / 128, CCH / 16, BB), blkG, 0, stream>>>(
        mlp_w2, mlp_wsc, CCH, CCH + 2 * CCH, mlp_b2, mlp_bsc,
        h1, CCH, 0, xcat, 2 * CCH, 0,
        CCH, NPTS, value_f, value_bf, nullptr);

    // 5-7) k, q, v projections        [OUTF]
    wmma_gemm_kernel<false,false,false,false,true,false>
        <<<dim3(NPTS / 128, DCH / 16, BB), blkG, 0, stream>>>(
        wk, nullptr, CCH, CCH, bk, nullptr,
        xcat, 2 * CCH, 0, nullptr, 0, 0,
        DCH, NPTS, kf, nullptr, nullptr);
    wmma_gemm_kernel<false,false,false,false,true,false>
        <<<dim3(NPTS / 128, DCH / 16, BB), blkG, 0, stream>>>(
        wq, nullptr, CCH, CCH, bq, nullptr,
        xcat, 2 * CCH, CCH, nullptr, 0, 0,
        DCH, NPTS, qf, nullptr, nullptr);
    wmma_gemm_kernel<false,false,false,false,true,false>
        <<<dim3(NPTS / 128, DCH / 16, BB), blkG, 0, stream>>>(
        wv, nullptr, CCH, CCH, bv, nullptr,
        value_bf, CCH, 0, nullptr, 0, 0,
        DCH, NPTS, vf, nullptr, nullptr);

    // 8) pos-MLP layer 1 (Cin=3, VALU) -> pre-BN hpos
    posmlp1_kernel<<<dim3((BB * NPTS * KNN) / 256), blk256, 0, stream>>>(
        pos, idx, pos_w1, pos_b1, hpos);

    // 9-10) BN stats + apply + ReLU on hpos
    bnstats_kernel<<<dim3(PHC), blk256, 0, stream>>>(hpos, pos_g, pos_beta, ss_pos, PHC, NKC);
    bnrelu_kernel<<<dim3((BB * PHC * NKC) / 256), blk256, 0, stream>>>(
        hpos, ss_pos, PHC, NKC, BB * PHC * NKC);

    // 11) pos_emb = pos_w2 . hpos + b2          [OUTF]
    wmma_gemm_kernel<false,false,false,false,true,false>
        <<<dim3(NKC / 128, DCH / 16, BB), blkG, 0, stream>>>(
        pos_w2, nullptr, PHC, PHC, pos_b2, nullptr,
        hpos, PHC, 0, nullptr, 0, 0,
        DCH, NKC, posemb, nullptr, nullptr);

    // 12) s = qk_rel + pos_emb (bf16)
    qkrel_kernel<<<dim3((unsigned)(((size_t)BB * DCH * NPTS * KNN) / 256)), blk256, 0, stream>>>(
        qf, kf, idx, posemb, s_bf);

    // 13) a_pre = attn_w1 . s + b1   (the 34-GFLOP GEMM)   [OUTH]
    wmma_gemm_kernel<false,false,false,false,false,true>
        <<<dim3(NKC / 128, AHC / 16, BB), blkG, 0, stream>>>(
        attn_w1, nullptr, DCH, DCH, attn_b1, nullptr,
        s_bf, DCH, 0, nullptr, 0, 0,
        AHC, NKC, nullptr, a_bf, nullptr);

    // 14-15) BN stats + apply + ReLU on a
    bnstats_kernel<<<dim3(AHC), blk256, 0, stream>>>(a_bf, attn_g, attn_beta, ss_attn, AHC, NKC);
    bnrelu_kernel<<<dim3((unsigned)(((size_t)BB * AHC * NKC) / 256)), blk256, 0, stream>>>(
        a_bf, ss_attn, AHC, NKC, (int)((size_t)BB * AHC * NKC));

    // 16) e = attn_w2 . a + b2   (pre-softmax logits)   [OUTF]
    wmma_gemm_kernel<false,false,false,false,true,false>
        <<<dim3(NKC / 128, DCH / 16, BB), blkG, 0, stream>>>(
        attn_w2, nullptr, AHC, AHC, attn_b2, nullptr,
        a_bf, AHC, 0, nullptr, 0, 0,
        DCH, NKC, e_f, nullptr, nullptr);

    // 17) softmax over K + aggregate with (v + pos_emb)
    softagg_kernel<<<dim3((BB * DCH * NPTS) / 256), blk256, 0, stream>>>(
        e_f, vf, posemb, agg_bf);

    // 18) y = end_w . agg + end_b + identity(value)   [RES, OUTF]
    wmma_gemm_kernel<false,false,true,false,true,false>
        <<<dim3(NPTS / 128, CCH / 16, BB), blkG, 0, stream>>>(
        end_w, nullptr, DCH, DCH, end_b, nullptr,
        agg_bf, DCH, 0, nullptr, 0, 0,
        CCH, NPTS, out, nullptr, value_f);
}
